// CancerGNN_61942018343011
// MI455X (gfx1250) — matrix operand order repaired and checked by reading, not compiled, not run on hardware
//
#include <hip/hip_runtime.h>

typedef __bf16    v16bf  __attribute__((ext_vector_type(16)));
typedef float     v8f    __attribute__((ext_vector_type(8)));
typedef unsigned  u32x4  __attribute__((ext_vector_type(4)));
typedef float     f32x4  __attribute__((ext_vector_type(4)));

#define N_NODES  50000
#define N_EDGES  800000
#define E_TOT    (N_EDGES + N_NODES)   // + self loops
#define IN_DIM   128
#define HID      64
#define HEADS    4
#define N_GRAPHS 8
#define NEG_SLOPE 0.2f

// ---------- helpers ----------
static __device__ __forceinline__ unsigned short f2bf(float f) {
  unsigned u = __float_as_uint(f);
  u += 0x7FFFu + ((u >> 16) & 1u);           // round-to-nearest-even
  return (unsigned short)(u >> 16);
}
// monotone float<->uint map so atomicMax(u32) == float max, and 0 == -inf
static __device__ __forceinline__ unsigned ford(float f) {
  unsigned u = __float_as_uint(f);
  return (u & 0x80000000u) ? ~u : (u | 0x80000000u);
}
static __device__ __forceinline__ float funord(unsigned u) {
  return __uint_as_float((u & 0x80000000u) ? (u & 0x7FFFFFFFu) : ~u);
}

// ---------- conversion / packing ----------
__global__ void cvt_bf16_k(const float* __restrict__ src,
                           unsigned short* __restrict__ dst, int n) {
  int t = blockIdx.x * blockDim.x + threadIdx.x;
  if (t < n) dst[t] = f2bf(src[t]);
}

// Pack f32 W[K][Ncols] (row-major) into the WMMA bf16 B-operand lane layout.
// Layout per 16x16x32 tile-step: u32 index ((nt*Ks+ks)*32 + lane)*8 + j holds
// K = ks*32 + 2j + (lane/16)*16 (+1 in high 16 bits), N = nt*16 + lane%16.
__global__ void pack_b_k(const float* __restrict__ W, unsigned* __restrict__ Bp,
                         int K, int Ncols) {
  int Ks = K >> 5, Ntn = Ncols >> 4;
  int total = Ntn * Ks * 32 * 8;
  int t = blockIdx.x * blockDim.x + threadIdx.x;
  if (t >= total) return;
  int j = t & 7, lane = (t >> 3) & 31, rest = t >> 8;
  int ks = rest % Ks, nt = rest / Ks;
  int half = lane >> 4, l16 = lane & 15;
  int n  = nt * 16 + l16;
  int k0 = ks * 32 + 2 * j + half * 16;
  unsigned lo = f2bf(W[(size_t)k0 * Ncols + n]);
  unsigned hi = f2bf(W[(size_t)(k0 + 1) * Ncols + n]);
  Bp[t] = lo | (hi << 16);
}

union BfVec { v16bf v; u32x4 q[2]; };

// ---------- WMMA GEMM: C[Mt*16 x Nt*16] = A_bf16(row-major, lda=K) @ Bp ----------
// A lane layout (ISA 7.12.2, 16-bit A 16x32): lane holds row M = lane%16;
// VGPR j: K = 2j+s + (lane/16)*16 (contiguous k order used here, same as Bp; not the ISA table order)
// => two contiguous 16B runs per K-step in a row-major bf16 row.
__global__ void gemm_bf16_wmma_k(const unsigned short* __restrict__ A,
                                 const unsigned* __restrict__ Bp,
                                 float* __restrict__ C,
                                 int Mt, int Nt, int Ks) {
  const int wave = threadIdx.x >> 5;
  const int lane = threadIdx.x & 31;
  const int tile = blockIdx.x * (blockDim.x >> 5) + wave;
  if (tile >= Mt * Nt) return;
  const int mt = tile / Nt, nt = tile % Nt;
  const int half = lane >> 4, l16 = lane & 15;
  const int K = Ks << 5, N = Nt << 4;
  const u32x4* arow = (const u32x4*)(A + (size_t)(mt * 16 + l16) * K);
  const u32x4* brow = (const u32x4*)(Bp + (size_t)nt * Ks * 256);
  v8f acc = {};
  for (int ks = 0; ks < Ks; ++ks) {
    BfVec a, b;
    a.q[0] = arow[ks * 4 + half * 2];     // K = ks*32 + half*16 + 0..7
    a.q[1] = arow[ks * 4 + half * 2 + 1]; // K = ks*32 + half*16 + 8..15
    b.q[0] = brow[(ks * 32 + lane) * 2];
    b.q[1] = brow[(ks * 32 + lane) * 2 + 1];
    acc = __builtin_amdgcn_wmma_f32_16x16x32_bf16(
        false, a.v, false, b.v, (short)0, acc, false, false);
  }
  // C/D layout: VGPR r -> M = r + half*8, N = lane%16
  float* crow = C + (size_t)(mt * 16) * N + nt * 16 + l16;
#pragma unroll
  for (int r = 0; r < 8; ++r) crow[(size_t)(r + half * 8) * N] = acc[r];
}

// ---------- GAT attention pieces ----------
__global__ void alpha_k(const float* __restrict__ h,
                        const float* __restrict__ a_src,
                        const float* __restrict__ a_dst,
                        float* __restrict__ as, float* __restrict__ ad,
                        int n, int Hh) {
  int t = blockIdx.x * blockDim.x + threadIdx.x;
  if (t >= n * Hh) return;
  int node = t / Hh, hh = t % Hh;
  const float* row = h + ((size_t)node * Hh + hh) * HID;
  const float* vs = a_src + hh * HID;
  const float* vd = a_dst + hh * HID;
  float s = 0.f, d = 0.f;
#pragma unroll 8
  for (int c = 0; c < HID; ++c) { float x = row[c]; s += x * vs[c]; d += x * vd[c]; }
  as[t] = s; ad[t] = d;
}

__global__ void edge_max_k(const int* __restrict__ ei,
                           const float* __restrict__ as, const float* __restrict__ ad,
                           unsigned* __restrict__ m, int Hh) {
  int t = blockIdx.x * blockDim.x + threadIdx.x;
  if (t >= E_TOT * Hh) return;
  int e = t / Hh, hh = t % Hh;
  int src, dst;
  if (e < N_EDGES) { src = ei[e]; dst = ei[N_EDGES + e]; } else { src = dst = e - N_EDGES; }
  float v = as[src * Hh + hh] + ad[dst * Hh + hh];
  v = v > 0.f ? v : NEG_SLOPE * v;
  atomicMax(&m[dst * Hh + hh], ford(v));
}

__global__ void edge_exp_k(const int* __restrict__ ei,
                           const float* __restrict__ as, const float* __restrict__ ad,
                           const unsigned* __restrict__ m, float* __restrict__ s,
                           float* __restrict__ pbuf, int Hh) {
  int t = blockIdx.x * blockDim.x + threadIdx.x;
  if (t >= E_TOT * Hh) return;
  int e = t / Hh, hh = t % Hh;
  int src, dst;
  if (e < N_EDGES) { src = ei[e]; dst = ei[N_EDGES + e]; } else { src = dst = e - N_EDGES; }
  float v = as[src * Hh + hh] + ad[dst * Hh + hh];
  v = v > 0.f ? v : NEG_SLOPE * v;
  float p = expf(v - funord(m[dst * Hh + hh]));
  pbuf[t] = p;
  atomicAdd(&s[dst * Hh + hh], p);
}

__global__ void pnorm_k(const int* __restrict__ ei, const float* __restrict__ s,
                        float* __restrict__ pbuf, int Hh) {
  int t = blockIdx.x * blockDim.x + threadIdx.x;
  if (t >= E_TOT * Hh) return;
  int e = t / Hh, hh = t % Hh;
  int dst = (e < N_EDGES) ? ei[N_EDGES + e] : (e - N_EDGES);
  pbuf[t] /= (s[dst * Hh + hh] + 1e-16f);
}

// one thread = (edge, head, 4-channel chunk); h rows + out rows are L2-resident
__global__ void scatter_k(const int* __restrict__ ei, const float* __restrict__ h,
                          const float* __restrict__ pbuf, float* __restrict__ out,
                          int Hh) {
  const int C4 = HID >> 2;
  int t = blockIdx.x * blockDim.x + threadIdx.x;
  if (t >= E_TOT * Hh * C4) return;
  int c4 = t % C4; int t2 = t / C4; int hh = t2 % Hh; int e = t2 / Hh;
  int src, dst;
  if (e < N_EDGES) { src = ei[e]; dst = ei[N_EDGES + e]; } else { src = dst = e - N_EDGES; }
  float w = pbuf[(size_t)e * Hh + hh];
  const f32x4 v = *(const f32x4*)(h + ((size_t)src * Hh + hh) * HID + c4 * 4);
  float* o = out + ((size_t)dst * Hh + hh) * HID + c4 * 4;
  atomicAdd(o + 0, w * v[0]);
  atomicAdd(o + 1, w * v[1]);
  atomicAdd(o + 2, w * v[2]);
  atomicAdd(o + 3, w * v[3]);
}

// out1 + b1 -> ELU -> bf16 activation for layer-2 GEMM
__global__ void finish1_k(const float* __restrict__ out1, const float* __restrict__ b1,
                          unsigned short* __restrict__ h1act) {
  int t = blockIdx.x * blockDim.x + threadIdx.x;
  if (t >= N_NODES * HEADS * HID) return;
  int c = t % (HEADS * HID);
  float v = out1[t] + b1[c];
  v = v > 0.f ? v : (expf(v) - 1.f);
  h1act[t] = f2bf(v);
}

// graph pooling: mean (sum+count) and max (monotone-uint) over batch segments
__global__ void pool_k(const float* __restrict__ out2, const float* __restrict__ b2,
                       const int* __restrict__ batch,
                       float* __restrict__ psum, unsigned* __restrict__ pmax,
                       float* __restrict__ pcnt) {
  int t = blockIdx.x * blockDim.x + threadIdx.x;
  if (t >= N_NODES * HID) return;
  int n = t / HID, c = t % HID;
  float v = out2[t] + b2[c];
  int b = batch[n];
  atomicAdd(&psum[b * HID + c], v);
  atomicMax(&pmax[b * HID + c], ford(v));
  if (c == 0) atomicAdd(&pcnt[b], 1.f);
}

// 8-row MLP head: one lane per graph (tiny)
__global__ void head_k(const float* __restrict__ psum, const unsigned* __restrict__ pmax,
                       const float* __restrict__ pcnt,
                       const float* __restrict__ cW1, const float* __restrict__ cb1,
                       const float* __restrict__ cW2, const float* __restrict__ cb2,
                       const float* __restrict__ cW3, const float* __restrict__ cb3,
                       float* __restrict__ out) {
  int b = threadIdx.x;
  if (b >= N_GRAPHS) return;
  float g[2 * HID];
  float cnt = fmaxf(pcnt[b], 1.0f);
  for (int i = 0; i < HID; ++i) {
    g[i]       = psum[b * HID + i] / cnt;
    g[HID + i] = funord(pmax[b * HID + i]);
  }
  float t1[HID];
  for (int j = 0; j < HID; ++j) {
    float acc = cb1[j];
    for (int i = 0; i < 2 * HID; ++i) acc += g[i] * cW1[i * HID + j];
    t1[j] = fmaxf(acc, 0.f);
  }
  float t2[32];
  for (int j = 0; j < 32; ++j) {
    float acc = cb2[j];
    for (int i = 0; i < HID; ++i) acc += t1[i] * cW2[i * 32 + j];
    t2[j] = fmaxf(acc, 0.f);
  }
  for (int j = 0; j < 2; ++j) {
    float acc = cb3[j];
    for (int i = 0; i < 32; ++i) acc += t2[i] * cW3[i * 2 + j];
    out[b * 2 + j] = acc;
  }
}

// ---------- host ----------
extern "C" void kernel_launch(void* const* d_in, const int* in_sizes, int n_in,
                              void* d_out, int out_size, void* d_ws, size_t ws_size,
                              hipStream_t stream) {
  (void)in_sizes; (void)n_in; (void)out_size; (void)ws_size;
  const float* x      = (const float*)d_in[0];
  const int*   ei     = (const int*)d_in[1];
  const int*   batch  = (const int*)d_in[2];
  const float* W1     = (const float*)d_in[3];
  const float* a_src1 = (const float*)d_in[4];
  const float* a_dst1 = (const float*)d_in[5];
  const float* b1     = (const float*)d_in[6];
  const float* W2     = (const float*)d_in[7];
  const float* a_src2 = (const float*)d_in[8];
  const float* a_dst2 = (const float*)d_in[9];
  const float* b2     = (const float*)d_in[10];
  const float* cW1    = (const float*)d_in[11];
  const float* cb1    = (const float*)d_in[12];
  const float* cW2    = (const float*)d_in[13];
  const float* cb2    = (const float*)d_in[14];
  const float* cW3    = (const float*)d_in[15];
  const float* cb3    = (const float*)d_in[16];
  float* out = (float*)d_out;

  // workspace carve-up (256B aligned)
  char* ws = (char*)d_ws;
  size_t off = 0;
  auto alloc = [&](size_t bytes) -> char* {
    char* p = ws + off;
    off += (bytes + 255) & ~(size_t)255;
    return p;
  };
  unsigned short* Xbf   = (unsigned short*)alloc((size_t)N_NODES * IN_DIM * 2);
  unsigned*       Bp1   = (unsigned*)alloc((size_t)IN_DIM * HEADS * HID * 2);
  unsigned*       Bp2   = (unsigned*)alloc((size_t)HEADS * HID * HID * 2);
  float*          h1    = (float*)alloc((size_t)N_NODES * HEADS * HID * 4);
  float*          as1   = (float*)alloc((size_t)N_NODES * HEADS * 4);
  float*          ad1   = (float*)alloc((size_t)N_NODES * HEADS * 4);
  unsigned*       m1    = (unsigned*)alloc((size_t)N_NODES * HEADS * 4);
  float*          s1    = (float*)alloc((size_t)N_NODES * HEADS * 4);
  float*          pb1   = (float*)alloc((size_t)E_TOT * HEADS * 4);
  float*          out1  = (float*)alloc((size_t)N_NODES * HEADS * HID * 4);
  unsigned short* h1act = (unsigned short*)alloc((size_t)N_NODES * HEADS * HID * 2);
  float*          h2    = (float*)alloc((size_t)N_NODES * HID * 4);
  float*          as2   = (float*)alloc((size_t)N_NODES * 4);
  float*          ad2   = (float*)alloc((size_t)N_NODES * 4);
  unsigned*       m2    = (unsigned*)alloc((size_t)N_NODES * 4);
  float*          s2    = (float*)alloc((size_t)N_NODES * 4);
  float*          pb2   = (float*)alloc((size_t)E_TOT * 4);
  float*          out2  = (float*)alloc((size_t)N_NODES * HID * 4);
  float*          psum  = (float*)alloc((size_t)N_GRAPHS * HID * 4);
  unsigned*       pmax  = (unsigned*)alloc((size_t)N_GRAPHS * HID * 4);
  float*          pcnt  = (float*)alloc((size_t)N_GRAPHS * 4);

  auto cdiv = [](long n, int b) { return (unsigned)((n + b - 1) / b); };
  const int B = 256;

  // zero accumulators (0 == -inf under the monotone uint max encoding)
  hipMemsetAsync(out1, 0, (size_t)N_NODES * HEADS * HID * 4, stream);
  hipMemsetAsync(m1,   0, (size_t)N_NODES * HEADS * 4, stream);
  hipMemsetAsync(s1,   0, (size_t)N_NODES * HEADS * 4, stream);
  hipMemsetAsync(out2, 0, (size_t)N_NODES * HID * 4, stream);
  hipMemsetAsync(m2,   0, (size_t)N_NODES * 4, stream);
  hipMemsetAsync(s2,   0, (size_t)N_NODES * 4, stream);
  hipMemsetAsync(psum, 0, (size_t)N_GRAPHS * HID * 4, stream);
  hipMemsetAsync(pmax, 0, (size_t)N_GRAPHS * HID * 4, stream);
  hipMemsetAsync(pcnt, 0, (size_t)N_GRAPHS * 4, stream);

  // convert + pack operands
  cvt_bf16_k<<<cdiv((long)N_NODES * IN_DIM, B), B, 0, stream>>>(x, Xbf, N_NODES * IN_DIM);
  pack_b_k<<<cdiv((long)(HEADS * HID / 16) * (IN_DIM / 32) * 256, B), B, 0, stream>>>(W1, Bp1, IN_DIM, HEADS * HID);
  pack_b_k<<<cdiv((long)(HID / 16) * (HEADS * HID / 32) * 256, B), B, 0, stream>>>(W2, Bp2, HEADS * HID, HID);

  // ---- layer 1 ----
  // h1 = X @ W1 : 3125x16 tiles, K=128 (4 wmma steps), 4 waves/block
  gemm_bf16_wmma_k<<<cdiv((long)3125 * 16, 4), 128, 0, stream>>>(Xbf, Bp1, h1, 3125, 16, IN_DIM / 32);
  alpha_k<<<cdiv((long)N_NODES * HEADS, B), B, 0, stream>>>(h1, a_src1, a_dst1, as1, ad1, N_NODES, HEADS);
  edge_max_k<<<cdiv((long)E_TOT * HEADS, B), B, 0, stream>>>(ei, as1, ad1, m1, HEADS);
  edge_exp_k<<<cdiv((long)E_TOT * HEADS, B), B, 0, stream>>>(ei, as1, ad1, m1, s1, pb1, HEADS);
  pnorm_k<<<cdiv((long)E_TOT * HEADS, B), B, 0, stream>>>(ei, s1, pb1, HEADS);
  scatter_k<<<cdiv((long)E_TOT * HEADS * (HID / 4), B), B, 0, stream>>>(ei, h1, pb1, out1, HEADS);
  finish1_k<<<cdiv((long)N_NODES * HEADS * HID, B), B, 0, stream>>>(out1, b1, h1act);

  // ---- layer 2 ----
  // h2 = elu(h1) @ W2 : 3125x4 tiles, K=256 (8 wmma steps)
  gemm_bf16_wmma_k<<<cdiv((long)3125 * 4, 4), 128, 0, stream>>>(h1act, Bp2, h2, 3125, 4, (HEADS * HID) / 32);
  alpha_k<<<cdiv((long)N_NODES, B), B, 0, stream>>>(h2, a_src2, a_dst2, as2, ad2, N_NODES, 1);
  edge_max_k<<<cdiv((long)E_TOT, B), B, 0, stream>>>(ei, as2, ad2, m2, 1);
  edge_exp_k<<<cdiv((long)E_TOT, B), B, 0, stream>>>(ei, as2, ad2, m2, s2, pb2, 1);
  pnorm_k<<<cdiv((long)E_TOT, B), B, 0, stream>>>(ei, s2, pb2, 1);
  scatter_k<<<cdiv((long)E_TOT * (HID / 4), B), B, 0, stream>>>(ei, h2, pb2, out2, 1);

  // ---- pooling + classifier head ----
  pool_k<<<cdiv((long)N_NODES * HID, B), B, 0, stream>>>(out2, b2, batch, psum, pmax, pcnt);
  head_k<<<1, 32, 0, stream>>>(psum, pmax, pcnt, cW1, cb1, cW2, cb2, cW3, cb3, out);
}
